// MultiheadAttention_49813030699093
// MI455X (gfx1250) — compile-verified, matrix operand order checked by reading, not run on hardware
//
#include <hip/hip_runtime.h>
#include <hip/hip_bf16.h>

// ---------------- problem constants ----------------
#define S    2048
#define E    1024
#define H    16
#define KVH  8
#define D    64
#define KVD  (KVH * D)      // 512
#define NTOT (E + 2 * KVD)  // 2048 combined qkv output cols
#define CLAMP_V 50000.0f
#define QTILE 16            // query rows per fused-attention block

typedef __attribute__((ext_vector_type(16))) __bf16 v16bf;
typedef __attribute__((ext_vector_type(8)))  float  v8f;

// ---------------- WMMA helpers ----------------
__device__ __forceinline__ v8f wmma_bf16(v16bf a, v16bf b, v8f c) {
  return __builtin_amdgcn_wmma_f32_16x16x32_bf16(
      false, a, false, b, (short)0, c, false, false);
}

// Per-lane fragment gather: 8 contiguous bf16 at p, 8 more at p+16.
// (A: lane row = L%16, K halves split at lane bit 4; B symmetric with cols.)
__device__ __forceinline__ v16bf frag_from(const __bf16* p) {
  v16bf f;
#pragma unroll
  for (int e = 0; e < 8; ++e) { f[e] = p[e]; f[e + 8] = p[e + 16]; }
  return f;
}

// Same gather but from f32 source with bf16 conversion.
__device__ __forceinline__ v16bf frag_from_f32(const float* p) {
  v16bf f;
#pragma unroll
  for (int e = 0; e < 8; ++e) { f[e] = (__bf16)p[e]; f[e + 8] = (__bf16)p[e + 16]; }
  return f;
}

// LDS-tile fragment (row-major, leading dim ld), rows r0..r0+15, K 0..31.
__device__ __forceinline__ v16bf frag16x32(const __bf16* tile, int ld, int r0) {
  const int lane = threadIdx.x & 31;
  return frag_from(tile + (r0 + (lane & 15)) * ld + ((lane & 16) ? 8 : 0));
}

// ---------------- kernel 1: fused QKV projection ----------------
__global__ __launch_bounds__(256) void qkv_kernel(
    const float* __restrict__ x,
    const float* __restrict__ wq,
    const float* __restrict__ wk, const float* __restrict__ wkb,
    const float* __restrict__ wv, const float* __restrict__ wvb,
    float* __restrict__ qkv) {
  __shared__ __bf16 As[128][40];
  __shared__ __bf16 Bs[64][40];
  const int t = threadIdx.x;
  const int n_blk = blockIdx.x * 64;
  const int m_blk = blockIdx.y * 128;

  const float* Wp; const float* bias; int nloc;
  if (n_blk < E)            { Wp = wq; bias = nullptr; nloc = n_blk; }
  else if (n_blk < E + KVD) { Wp = wk; bias = wkb;     nloc = n_blk - E; }
  else                      { Wp = wv; bias = wvb;     nloc = n_blk - (E + KVD); }

  const int w = t >> 5, wm = w >> 1, wn = w & 1;
  v8f acc[2][2] = {};

  for (int k0 = 0; k0 < E; k0 += 32) {
    { const int r = t >> 1, cs = (t & 1) * 16;
      const float* src = x + (size_t)(m_blk + r) * E + k0 + cs;
#pragma unroll
      for (int i = 0; i < 16; i += 4) {
        float4 v = *(const float4*)(src + i);
        As[r][cs + i + 0] = (__bf16)v.x; As[r][cs + i + 1] = (__bf16)v.y;
        As[r][cs + i + 2] = (__bf16)v.z; As[r][cs + i + 3] = (__bf16)v.w;
      } }
    { const int r = t >> 2, cs = (t & 3) * 8;
      const float* src = Wp + (size_t)(nloc + r) * E + k0 + cs;
#pragma unroll
      for (int i = 0; i < 8; i += 4) {
        float4 v = *(const float4*)(src + i);
        Bs[r][cs + i + 0] = (__bf16)v.x; Bs[r][cs + i + 1] = (__bf16)v.y;
        Bs[r][cs + i + 2] = (__bf16)v.z; Bs[r][cs + i + 3] = (__bf16)v.w;
      } }
    __syncthreads();
    v16bf a0 = frag16x32(&As[0][0], 40, wm * 32);
    v16bf a1 = frag16x32(&As[0][0], 40, wm * 32 + 16);
    v16bf b0 = frag16x32(&Bs[0][0], 40, wn * 32);
    v16bf b1 = frag16x32(&Bs[0][0], 40, wn * 32 + 16);
    acc[0][0] = wmma_bf16(a0, b0, acc[0][0]);
    acc[0][1] = wmma_bf16(a0, b1, acc[0][1]);
    acc[1][0] = wmma_bf16(a1, b0, acc[1][0]);
    acc[1][1] = wmma_bf16(a1, b1, acc[1][1]);
    __syncthreads();
  }

  const int lane = t & 31;
#pragma unroll
  for (int i = 0; i < 2; ++i)
#pragma unroll
    for (int j = 0; j < 2; ++j) {
      const int gn = n_blk + wn * 32 + j * 16 + (lane & 15);
      const int bn = nloc  + wn * 32 + j * 16 + (lane & 15);
      const float bv = bias ? bias[bn] : 0.0f;
      const int mb = m_blk + wm * 32 + i * 16 + ((lane & 16) ? 8 : 0);
#pragma unroll
      for (int v = 0; v < 8; ++v)
        qkv[(size_t)(mb + v) * NTOT + gn] = acc[i][j][v] + bv;
    }
}

// ---------------- kernel 2: RoPE + pack Q,K to bf16 [head][s][d] ----------------
__global__ __launch_bounds__(256) void rope_pack_kernel(
    const float* __restrict__ qkv,
    const float* __restrict__ fc, const float* __restrict__ fs,
    __bf16* __restrict__ Qb, __bf16* __restrict__ Kb) {
  const int QP = S * H * (D / 2);
  const int KP = S * KVH * (D / 2);
  int idx = blockIdx.x * blockDim.x + threadIdx.x;
  if (idx < QP) {
    const int i = idx & 31, h = (idx >> 5) & 15, s = idx >> 9;
    const float* src = qkv + (size_t)s * NTOT + h * D + 2 * i;
    const float xr = src[0], xi = src[1];
    const float c = fc[s * 32 + i], sn = fs[s * 32 + i];
    __bf16* dst = Qb + ((size_t)h * S + s) * D + 2 * i;
    dst[0] = (__bf16)(xr * c - xi * sn);
    dst[1] = (__bf16)(xr * sn + xi * c);
  } else if (idx < QP + KP) {
    const int l = idx - QP;
    const int i = l & 31, h = (l >> 5) & 7, s = l >> 8;
    const float* src = qkv + (size_t)s * NTOT + E + h * D + 2 * i;
    const float xr = src[0], xi = src[1];
    const float c = fc[s * 32 + i], sn = fs[s * 32 + i];
    __bf16* dst = Kb + ((size_t)h * S + s) * D + 2 * i;
    dst[0] = (__bf16)(xr * c - xi * sn);
    dst[1] = (__bf16)(xr * sn + xi * c);
  }
}

// ---------------- kernel 3: pack V transposed -> Vt[kv][d][s] (bf16) ----------------
__global__ __launch_bounds__(256) void vpack_kernel(
    const float* __restrict__ qkv, __bf16* __restrict__ Vt) {
  const int idx = blockIdx.x * blockDim.x + threadIdx.x;  // 2^17 threads
  const int s8 = idx & 255, d = (idx >> 8) & 63, kv = idx >> 14;
  const int s0 = s8 * 8;
  const float* src = qkv + (size_t)s0 * NTOT + E + KVD + kv * D + d;  // stride NTOT (L2-resident)
  __bf16* dst = Vt + ((size_t)kv * D + d) * S + s0;
#pragma unroll
  for (int j = 0; j < 8; ++j) dst[j] = (__bf16)src[(size_t)j * NTOT];
}

// ---------------- kernel 4: fused logits + softmax + P@V ----------------
// One block: 16 query rows x one head. Logit stripe (16x2048 f32 = 128KB) lives in LDS.
__global__ __launch_bounds__(256) void fused_attn_kernel(
    const __bf16* __restrict__ Qb, const __bf16* __restrict__ Kb,
    const __bf16* __restrict__ Vt, const float* __restrict__ bias,
    float* __restrict__ attn, float* __restrict__ ctx) {
  extern __shared__ char smem[];
  float* Pbuf  = (float*)smem;                         // [16][2048]  131072 B
  float* Red   = (float*)(smem + 131072);              // [8][16][16]   8192 B
  float* statM = (float*)(smem + 131072 + 8192);       // [256]
  float* statS = (float*)(smem + 131072 + 8192 + 1024);// [256]
  float* smInv = (float*)(smem + 131072 + 8192 + 2048);// [16]

  const int h = blockIdx.y, kv = h >> 1;
  const int q0 = blockIdx.x * QTILE;
  const __bf16* Qp = Qb + (size_t)h  * S * D;
  const __bf16* Kp = Kb + (size_t)kv * S * D;
  const __bf16* Vp = Vt + (size_t)kv * D * S;
  const int t = threadIdx.x, w = t >> 5, lane = t & 31;
  const int lrow = lane & 15, rbase = (lane & 16) ? 8 : 0;

  // -------- phase 1: logits for keys [w*256, w*256+256) --------
  const __bf16* qrow = Qp + (size_t)(q0 + lrow) * D + rbase;  // rbase doubles as K-half offset
  const v16bf a0 = frag_from(qrow);
  const v16bf a1 = frag_from(qrow + 32);

  for (int kt = w * 256; kt < w * 256 + 256; kt += 16) {
    const __bf16* krow = Kp + (size_t)(kt + lrow) * D + rbase;
    __builtin_prefetch(Kp + (size_t)(kt + 16 + lrow) * D, 0, 1);
    v16bf b0 = frag_from(krow);
    v16bf b1 = frag_from(krow + 32);
    v8f acc = {};
    acc = wmma_bf16(a0, b0, acc);
    acc = wmma_bf16(a1, b1, acc);
    const int col = kt + lrow;
#pragma unroll
    for (int v = 0; v < 8; ++v) {
      const int r = v + rbase;
      float val = acc[v] * 0.125f;
      val = fminf(fmaxf(val, -CLAMP_V), CLAMP_V);
      val += bias[((size_t)h * S + q0 + r) * S + col];
      Pbuf[r * 2048 + col] = val;
    }
  }
  __syncthreads();

  // -------- phase 2: softmax over each of the 16 rows --------
  const int r = t >> 4, c = t & 15;
  float m = -3.0e38f;
  for (int j = c; j < 2048; j += 16) m = fmaxf(m, Pbuf[r * 2048 + j]);
  statM[r * 16 + c] = m;
  __syncthreads();
  float rm = statM[r * 16];
#pragma unroll
  for (int i = 1; i < 16; ++i) rm = fmaxf(rm, statM[r * 16 + i]);
  float s = 0.0f;
  for (int j = c; j < 2048; j += 16) {
    const float e = __expf(Pbuf[r * 2048 + j] - rm);
    Pbuf[r * 2048 + j] = e;  // keep unnormalized exp in LDS
    s += e;
  }
  statS[r * 16 + c] = s;
  __syncthreads();
  float tot = 0.0f;
#pragma unroll
  for (int i = 0; i < 16; ++i) tot += statS[r * 16 + i];
  if (c == 0) smInv[r] = 1.0f / tot;
  __syncthreads();

  // coalesced write of normalized attention weights (output #2)
  for (int rr = 0; rr < QTILE; ++rr) {
    const float iv = smInv[rr];
    float* dst = attn + ((size_t)h * S + q0 + rr) * S;
    for (int j = t; j < 2048; j += 256) dst[j] = Pbuf[rr * 2048 + j] * iv;
  }

  // -------- phase 3: P @ V  (wave w -> d-tile w&3, key-half w>>2) --------
  const int dtile = w & 3, khalf = w >> 2;
  const __bf16* vcol = Vp + (size_t)(dtile * 16 + lrow) * S + rbase;
  v8f acc = {};
  for (int k0 = khalf * 1024; k0 < khalf * 1024 + 1024; k0 += 32) {
    v16bf a = frag_from_f32(Pbuf + lrow * 2048 + k0 + rbase);
    v16bf b = frag_from(vcol + k0);
    acc = wmma_bf16(a, b, acc);
  }
#pragma unroll
  for (int v = 0; v < 8; ++v)
    Red[((w * 16) + v + rbase) * 16 + lrow] = acc[v];
  __syncthreads();

  // cross-wave reduce (2 key-halves) + normalize + write context
  for (int e = t; e < QTILE * D; e += 256) {
    const int rr = e >> 6, d = e & 63, dt = d >> 4, dc = d & 15;
    const float val = (Red[((dt * 16) + rr) * 16 + dc] +
                       Red[(((dt + 4) * 16) + rr) * 16 + dc]) * smInv[rr];
    ctx[(size_t)(q0 + rr) * E + h * D + d] = val;
  }
}

// ---------------- kernel 5: out = ctx @ Wo^T + b ----------------
__global__ __launch_bounds__(256) void oproj_kernel(
    const float* __restrict__ ctx, const float* __restrict__ wo,
    const float* __restrict__ wob, float* __restrict__ out) {
  __shared__ __bf16 As[128][40];
  __shared__ __bf16 Bs[64][40];
  const int t = threadIdx.x;
  const int n_blk = blockIdx.x * 64;
  const int m_blk = blockIdx.y * 128;
  const int w = t >> 5, wm = w >> 1, wn = w & 1;
  v8f acc[2][2] = {};

  for (int k0 = 0; k0 < E; k0 += 32) {
    { const int r = t >> 1, cs = (t & 1) * 16;
      const float* src = ctx + (size_t)(m_blk + r) * E + k0 + cs;
#pragma unroll
      for (int i = 0; i < 16; i += 4) {
        float4 v = *(const float4*)(src + i);
        As[r][cs + i + 0] = (__bf16)v.x; As[r][cs + i + 1] = (__bf16)v.y;
        As[r][cs + i + 2] = (__bf16)v.z; As[r][cs + i + 3] = (__bf16)v.w;
      } }
    { const int r = t >> 2, cs = (t & 3) * 8;
      const float* src = wo + (size_t)(n_blk + r) * E + k0 + cs;
#pragma unroll
      for (int i = 0; i < 8; i += 4) {
        float4 v = *(const float4*)(src + i);
        Bs[r][cs + i + 0] = (__bf16)v.x; Bs[r][cs + i + 1] = (__bf16)v.y;
        Bs[r][cs + i + 2] = (__bf16)v.z; Bs[r][cs + i + 3] = (__bf16)v.w;
      } }
    __syncthreads();
    v16bf a0 = frag16x32(&As[0][0], 40, wm * 32);
    v16bf a1 = frag16x32(&As[0][0], 40, wm * 32 + 16);
    v16bf b0 = frag16x32(&Bs[0][0], 40, wn * 32);
    v16bf b1 = frag16x32(&Bs[0][0], 40, wn * 32 + 16);
    acc[0][0] = wmma_bf16(a0, b0, acc[0][0]);
    acc[0][1] = wmma_bf16(a0, b1, acc[0][1]);
    acc[1][0] = wmma_bf16(a1, b0, acc[1][0]);
    acc[1][1] = wmma_bf16(a1, b1, acc[1][1]);
    __syncthreads();
  }

  const int lane = t & 31;
#pragma unroll
  for (int i = 0; i < 2; ++i)
#pragma unroll
    for (int j = 0; j < 2; ++j) {
      const int gn = n_blk + wn * 32 + j * 16 + (lane & 15);
      const int mb = m_blk + wm * 32 + i * 16 + ((lane & 16) ? 8 : 0);
#pragma unroll
      for (int v = 0; v < 8; ++v)
        out[(size_t)(mb + v) * E + gn] = acc[i][j][v] + wob[gn];
    }
}

// ---------------- launch ----------------
extern "C" void kernel_launch(void* const* d_in, const int* in_sizes, int n_in,
                              void* d_out, int out_size, void* d_ws, size_t ws_size,
                              hipStream_t stream) {
  (void)in_sizes; (void)n_in; (void)out_size; (void)ws_size;
  const float* x   = (const float*)d_in[0];
  const float* ab  = (const float*)d_in[1];
  const float* fc  = (const float*)d_in[2];
  const float* fs  = (const float*)d_in[3];
  const float* wq  = (const float*)d_in[4];
  const float* wk  = (const float*)d_in[5];
  const float* wkb = (const float*)d_in[6];
  const float* wv  = (const float*)d_in[7];
  const float* wvb = (const float*)d_in[8];
  const float* wo  = (const float*)d_in[9];
  const float* wob = (const float*)d_in[10];

  float* out  = (float*)d_out;
  float* attn = out + (size_t)S * E;

  char* ws = (char*)d_ws;  // 32 MB used
  float*  qkv = (float*)(ws);                        // S*2048 f32 = 16 MB
  __bf16* Qb  = (__bf16*)(ws + 16u * 1024 * 1024);   // H*S*D     =  4 MB
  __bf16* Kb  = (__bf16*)(ws + 20u * 1024 * 1024);   // KVH*S*D   =  2 MB
  __bf16* Vt  = (__bf16*)(ws + 22u * 1024 * 1024);   // KVH*D*S   =  2 MB
  float*  ctx = (float*)(ws + 24u * 1024 * 1024);    // S*E f32   =  8 MB

  const size_t fused_lds = 131072 + 8192 + 2048 + 64;  // 141376 B

  qkv_kernel<<<dim3(NTOT / 64, S / 128), 256, 0, stream>>>(x, wq, wk, wkb, wv, wvb, qkv);
  rope_pack_kernel<<<dim3((S * E / 2 + S * KVD / 2) / 256), 256, 0, stream>>>(qkv, fc, fs, Qb, Kb);
  vpack_kernel<<<dim3((KVH * D * (S / 8)) / 256), 256, 0, stream>>>(qkv, Vt);
  fused_attn_kernel<<<dim3(S / QTILE, H), 256, fused_lds, stream>>>(Qb, Kb, Vt, ab, attn, ctx);
  oproj_kernel<<<dim3(E / 64, S / 128), 256, 0, stream>>>(ctx, wo, wob, out);
}
